// SelfAttention_67138928771419
// MI455X (gfx1250) — compile-verified, matrix operand order checked by reading, not run on hardware
//
#include <hip/hip_runtime.h>

// ---------------------------------------------------------------------------
// CDNA5 (gfx1250) fused self-attention: QKV GEMM + flash-style attention.
// All matrix math through v_wmma_f32_16x16x32_f16 (f16 in, f32 accumulate).
// ---------------------------------------------------------------------------

typedef __attribute__((ext_vector_type(16))) _Float16 v16h;
typedef __attribute__((ext_vector_type(8)))  _Float16 v8h;
typedef __attribute__((ext_vector_type(8)))  float    v8f;

#define BATCH   8
#define SEQ     2048
#define DIM     1024
#define NTRIPLE 3072

union HFrag { v16h v; v8h h[2]; };

__device__ __forceinline__ v8f wmma16x16x32(v16h a, v16h b, v8f c) {
  // (neg_a, A, neg_b, B, c_mod, C, reuse_a, reuse_b)
  return __builtin_amdgcn_wmma_f32_16x16x32_f16(false, a, false, b, (short)0, c,
                                                false, false);
}

__device__ __forceinline__ v8f v8f_zero() {
  v8f z;
#pragma unroll
  for (int i = 0; i < 8; ++i) z[i] = 0.0f;
  return z;
}

// ---------------------------------------------------------------------------
// Kernel 1: fp32 -> f16 conversion (x)
// ---------------------------------------------------------------------------
__global__ __launch_bounds__(256) void k_cvt_f16(const float* __restrict__ src,
                                                 _Float16* __restrict__ dst,
                                                 int n) {
  int i = blockIdx.x * 256 + threadIdx.x;
  if (i < n) dst[i] = (_Float16)src[i];
}

// ---------------------------------------------------------------------------
// Kernel 2: W [K=1024][N=3072] fp32 -> Wt [N=3072][K=1024] f16
// (makes WMMA B-operand fragment loads contiguous 16B reads)
// ---------------------------------------------------------------------------
__global__ __launch_bounds__(256) void k_transpose_w(const float* __restrict__ W,
                                                     _Float16* __restrict__ wt) {
  int i = blockIdx.x * 256 + threadIdx.x;  // i = n*1024 + k, exact coverage
  int n = i >> 10;
  int k = i & 1023;
  wt[i] = (_Float16)W[(size_t)k * NTRIPLE + n];
}

// ---------------------------------------------------------------------------
// Kernel 3: QKV projection  C[16384,3072] = xh @ Wt^T + bias, split into q/k/v
// Block: 128 threads = 4 waves (2x2), wave tile 64x32 = 4x2 WMMA frags.
// L2-resident operands -> direct global fragment loads, no LDS staging.
// ---------------------------------------------------------------------------
__global__ __launch_bounds__(128) void k_qkv_gemm(
    const _Float16* __restrict__ xh, const _Float16* __restrict__ wt,
    const float* __restrict__ bias, _Float16* __restrict__ qh,
    _Float16* __restrict__ kh, _Float16* __restrict__ vh) {
  const int lane = threadIdx.x & 31;
  const int wave = threadIdx.x >> 5;
  const int llo  = lane & 15;   // N (or M) index within fragment
  const int lhi  = lane >> 4;   // selects K-half of the fragment
  const int row0 = blockIdx.y * 128 + (wave >> 1) * 64;
  const int col0 = blockIdx.x * 64 + (wave & 1) * 32;

  v8f acc[4][2];
#pragma unroll
  for (int r = 0; r < 4; ++r)
#pragma unroll
    for (int c = 0; c < 2; ++c) acc[r][c] = v8f_zero();

  for (int k0 = 0; k0 < DIM; k0 += 32) {
    HFrag a[4], bm[2];
#pragma unroll
    for (int r = 0; r < 4; ++r) {
      const _Float16* p = xh + (size_t)(row0 + r * 16 + llo) * DIM + k0 + lhi * 8;
      a[r].h[0] = *(const v8h*)p;
      a[r].h[1] = *(const v8h*)(p + 16);
    }
#pragma unroll
    for (int c = 0; c < 2; ++c) {
      const _Float16* p = wt + (size_t)(col0 + c * 16 + llo) * DIM + k0 + lhi * 8;
      bm[c].h[0] = *(const v8h*)p;
      bm[c].h[1] = *(const v8h*)(p + 16);
    }
#pragma unroll
    for (int r = 0; r < 4; ++r)
#pragma unroll
      for (int c = 0; c < 2; ++c)
        acc[r][c] = wmma16x16x32(a[r].v, bm[c].v, acc[r][c]);
  }

  // Epilogue: add bias (fp32), convert to f16, scatter into q/k/v.
#pragma unroll
  for (int r = 0; r < 4; ++r)
#pragma unroll
    for (int c = 0; c < 2; ++c)
#pragma unroll
      for (int i = 0; i < 8; ++i) {
        int grow = row0 + r * 16 + i + 8 * lhi;  // C layout: row = i + 8*(lane/16)
        int gcol = col0 + c * 16 + llo;          //           col = lane%16
        float v = acc[r][c][i] + bias[gcol];
        int which = gcol >> 10;
        int cc = gcol & 1023;
        _Float16* dst = (which == 0) ? qh : ((which == 1) ? kh : vh);
        dst[(size_t)grow * DIM + cc] = (_Float16)v;
      }
}

// ---------------------------------------------------------------------------
// Kernel 4: softmax stats pass. Per (b, query-tile 64): each wave owns 16 rows,
// streams 32-key tiles, scores via WMMA over D, online max/sum via shfl_xor
// within the 16-lane N-groups of the wave32 fragment layout.
// ---------------------------------------------------------------------------
__global__ __launch_bounds__(128) void k_attn_stats(
    const _Float16* __restrict__ qh, const _Float16* __restrict__ kh,
    float* __restrict__ mrow, float* __restrict__ lrow,
    const int* __restrict__ mask_flag) {
  const bool causal = (mask_flag[0] != 0);
  const int lane = threadIdx.x & 31;
  const int wave = threadIdx.x >> 5;
  const int llo = lane & 15;
  const int lhi = lane >> 4;
  const int batch = blockIdx.y;
  const int q0 = blockIdx.x * 64 + wave * 16;
  const size_t qbase = (size_t)batch * SEQ;
  const float sc = 0.03125f;  // D^-0.5 = 1/32

  float m_run[8], l_run[8];
#pragma unroll
  for (int i = 0; i < 8; ++i) { m_run[i] = -3.0e38f; l_run[i] = 0.0f; }

  const int ntiles = causal ? (blockIdx.x * 2 + 2) : (SEQ / 32);

  for (int kt = 0; kt < ntiles; ++kt) {
    const int s0 = kt * 32;
    v8f sacc[2];
    sacc[0] = v8f_zero();
    sacc[1] = v8f_zero();
    for (int k0 = 0; k0 < DIM; k0 += 32) {
      HFrag a, b0, b1;
      const _Float16* pa = qh + (qbase + q0 + llo) * DIM + k0 + lhi * 8;
      a.h[0] = *(const v8h*)pa;
      a.h[1] = *(const v8h*)(pa + 16);
      const _Float16* pb0 = kh + (qbase + s0 + llo) * DIM + k0 + lhi * 8;
      b0.h[0] = *(const v8h*)pb0;
      b0.h[1] = *(const v8h*)(pb0 + 16);
      const _Float16* pb1 = pb0 + (size_t)16 * DIM;
      b1.h[0] = *(const v8h*)pb1;
      b1.h[1] = *(const v8h*)(pb1 + 16);
      sacc[0] = wmma16x16x32(a.v, b0.v, sacc[0]);
      sacc[1] = wmma16x16x32(a.v, b1.v, sacc[1]);
    }
#pragma unroll
    for (int i = 0; i < 8; ++i) {
      int qg = q0 + i + 8 * lhi;
      float s0v = sacc[0][i] * sc;
      float s1v = sacc[1][i] * sc;
      if (causal && (s0 + llo) > qg) s0v = -3.0e38f;
      if (causal && (s0 + 16 + llo) > qg) s1v = -3.0e38f;
      float mx = fmaxf(s0v, s1v);
      mx = fmaxf(mx, __shfl_xor(mx, 1));
      mx = fmaxf(mx, __shfl_xor(mx, 2));
      mx = fmaxf(mx, __shfl_xor(mx, 4));
      mx = fmaxf(mx, __shfl_xor(mx, 8));
      float mn = fmaxf(m_run[i], mx);
      float se = __expf(s0v - mn) + __expf(s1v - mn);
      se += __shfl_xor(se, 1);
      se += __shfl_xor(se, 2);
      se += __shfl_xor(se, 4);
      se += __shfl_xor(se, 8);
      l_run[i] = l_run[i] * __expf(m_run[i] - mn) + se;
      m_run[i] = mn;
    }
  }

  if (llo == 0) {  // lane 0 writes rows 0..7 of its wave tile, lane 16 rows 8..15
#pragma unroll
    for (int i = 0; i < 8; ++i) {
      size_t r = qbase + q0 + i + 8 * lhi;
      mrow[r] = m_run[i];
      lrow[r] = l_run[i];
    }
  }
}

// ---------------------------------------------------------------------------
// Kernel 5: output pass. Grid (d-tile 256, q-tile 64, batch). Per wave:
// 16 q-rows x 256 d-cols of O in registers (16 frags). S recomputed per d-tile
// via WMMA; P transposed through per-wave LDS into A-operand layout; V tile
// staged transposed in LDS so PV B-fragments are contiguous ds_load_b128.
// ---------------------------------------------------------------------------
__global__ __launch_bounds__(128) void k_attn_out(
    const _Float16* __restrict__ qh, const _Float16* __restrict__ kh,
    const _Float16* __restrict__ vh, const float* __restrict__ mrow,
    const float* __restrict__ lrow, float* __restrict__ out,
    const int* __restrict__ mask_flag) {
  __shared__ _Float16 sVt[256 * 40];     // V^T tile [d 256][key 32], stride 40 (16B aligned rows)
  __shared__ _Float16 sP[4 * 16 * 32];   // per-wave P [16 rows][32 keys]

  const bool causal = (mask_flag[0] != 0);
  const int lane = threadIdx.x & 31;
  const int wave = threadIdx.x >> 5;
  const int llo = lane & 15;
  const int lhi = lane >> 4;
  const int dt0 = blockIdx.x * 256;
  const int q0 = blockIdx.y * 64 + wave * 16;
  const int batch = blockIdx.z;
  const size_t qbase = (size_t)batch * SEQ;
  const float sc = 0.03125f;

  float m_i[8], l_i[8];
#pragma unroll
  for (int i = 0; i < 8; ++i) {
    size_t r = qbase + q0 + i + 8 * lhi;
    m_i[i] = mrow[r];
    l_i[i] = lrow[r];
  }

  v8f oacc[16];
#pragma unroll
  for (int j = 0; j < 16; ++j) oacc[j] = v8f_zero();

  const int ntiles = causal ? (blockIdx.y * 2 + 2) : (SEQ / 32);
  const int tkey = threadIdx.x & 31;    // staging: key owned by this thread
  const int tchunk = threadIdx.x >> 5;  // staging: 64-wide d chunk

  for (int kt = 0; kt < ntiles; ++kt) {
    const int s0 = kt * 32;

    // ---- stage V^T tile: V[key][d] (contiguous reads) -> sVt[d][key]
    {
      const _Float16* vsrc = vh + (qbase + s0 + tkey) * DIM + dt0 + tchunk * 64;
#pragma unroll
      for (int jv = 0; jv < 8; ++jv) {
        v8h chunk = *(const v8h*)(vsrc + jv * 8);
#pragma unroll
        for (int e = 0; e < 8; ++e)
          sVt[(tchunk * 64 + jv * 8 + e) * 40 + tkey] = chunk[e];
      }
    }
    __syncthreads();

    // ---- S = Q K^T (16 x 32 per wave), full-D WMMA chain
    v8f sacc[2];
    sacc[0] = v8f_zero();
    sacc[1] = v8f_zero();
    for (int k0 = 0; k0 < DIM; k0 += 32) {
      HFrag a, b0, b1;
      const _Float16* pa = qh + (qbase + q0 + llo) * DIM + k0 + lhi * 8;
      a.h[0] = *(const v8h*)pa;
      a.h[1] = *(const v8h*)(pa + 16);
      const _Float16* pb0 = kh + (qbase + s0 + llo) * DIM + k0 + lhi * 8;
      b0.h[0] = *(const v8h*)pb0;
      b0.h[1] = *(const v8h*)(pb0 + 16);
      const _Float16* pb1 = pb0 + (size_t)16 * DIM;
      b1.h[0] = *(const v8h*)pb1;
      b1.h[1] = *(const v8h*)(pb1 + 16);
      sacc[0] = wmma16x16x32(a.v, b0.v, sacc[0]);
      sacc[1] = wmma16x16x32(a.v, b1.v, sacc[1]);
    }

    // ---- P = exp(S*sc - m), causal-masked; C-layout -> A-layout via LDS
    _Float16* pw = sP + wave * (16 * 32);
#pragma unroll
    for (int i = 0; i < 8; ++i) {
      int qg = q0 + i + 8 * lhi;
      int row = i + 8 * lhi;
      float p0 = (causal && (s0 + llo) > qg) ? 0.0f
                                             : __expf(sacc[0][i] * sc - m_i[i]);
      float p1 = (causal && (s0 + 16 + llo) > qg)
                     ? 0.0f
                     : __expf(sacc[1][i] * sc - m_i[i]);
      pw[row * 32 + llo] = (_Float16)p0;
      pw[row * 32 + 16 + llo] = (_Float16)p1;
    }
    // Wave-local visibility: same-wave LDS ops are in-order, but make the
    // store->load hazard explicit with a CDNA5 split-counter wait.
    asm volatile("s_wait_dscnt 0" ::: "memory");

    HFrag pf;
    {
      const _Float16* pp = pw + llo * 32 + lhi * 8;
      pf.h[0] = *(const v8h*)pp;
      pf.h[1] = *(const v8h*)(pp + 16);
    }

    // ---- O += P V over this key tile (K-dim = 32 -> one WMMA per d-frag)
#pragma unroll
    for (int df = 0; df < 16; ++df) {
      HFrag vf;
      const _Float16* pv = sVt + (df * 16 + llo) * 40 + lhi * 8;
      vf.h[0] = *(const v8h*)pv;
      vf.h[1] = *(const v8h*)(pv + 16);
      oacc[df] = wmma16x16x32(pf.v, vf.v, oacc[df]);
    }
    __syncthreads();  // protect sVt before next tile's staging
  }

  // ---- normalize and write fp32 output
#pragma unroll
  for (int df = 0; df < 16; ++df)
#pragma unroll
    for (int i = 0; i < 8; ++i) {
      int qg = q0 + i + 8 * lhi;
      int d = dt0 + df * 16 + llo;
      out[(qbase + qg) * DIM + d] = oacc[df][i] / l_i[i];
    }
}

// ---------------------------------------------------------------------------
// Host launcher
// ---------------------------------------------------------------------------
extern "C" void kernel_launch(void* const* d_in, const int* in_sizes, int n_in,
                              void* d_out, int out_size, void* d_ws,
                              size_t ws_size, hipStream_t stream) {
  (void)in_sizes; (void)n_in; (void)out_size; (void)ws_size;
  const float* x = (const float*)d_in[0];
  const float* W = (const float*)d_in[1];
  const float* b = (const float*)d_in[2];
  const int* mask_flag = (const int*)d_in[3];
  float* out = (float*)d_out;

  char* ws = (char*)d_ws;
  // Workspace layout (bytes, all 16B aligned):
  _Float16* xh = (_Float16*)(ws + 0);           // 16384*1024 f16 = 32 MB
  _Float16* wt = (_Float16*)(ws + 33554432);    // 3072*1024 f16  =  6 MB
  _Float16* qh = (_Float16*)(ws + 39845888);    // 32 MB
  _Float16* kh = (_Float16*)(ws + 73400320);    // 32 MB
  _Float16* vh = (_Float16*)(ws + 106954752);   // 32 MB
  float* mrow = (float*)(ws + 140509184);       // 16384 f32
  float* lrow = (float*)(ws + 140574720);       // 16384 f32

  k_cvt_f16<<<65536, 256, 0, stream>>>(x, xh, BATCH * SEQ * DIM);
  k_transpose_w<<<12288, 256, 0, stream>>>(W, wt);
  k_qkv_gemm<<<dim3(48, 128), 128, 0, stream>>>(xh, wt, b, qh, kh, vh);
  k_attn_stats<<<dim3(32, 8), 128, 0, stream>>>(qh, kh, mrow, lrow, mask_flag);
  k_attn_out<<<dim3(4, 32, 8), 128, 0, stream>>>(qh, kh, vh, mrow, lrow, out,
                                                 mask_flag);
}
